// Crossatten_align_atttrans_17755394801959
// MI455X (gfx1250) — matrix operand order repaired and checked by reading, not compiled, not run on hardware
//
#include <hip/hip_runtime.h>
#include <hip/hip_bf16.h>
#include <math.h>

typedef __attribute__((ext_vector_type(16))) _Float16 v16h;
typedef __attribute__((ext_vector_type(8)))  float    v8f;

#define CDIM   192
#define NHEADS 6
#define DHEAD  32
#define NPIX   49          // 7*7 window
#define IMG    224
#define NWIN   8192        // 8 * 32 * 32
#define MROWS  401408      // NWIN * NPIX
#define SHIFTX 3
#define OUT_ELEMS 77070336 // 8*224*224*192

// ---------------- helpers ----------------

static __device__ __forceinline__ void cp32B(_Float16* __restrict__ dst,
                                             const _Float16* __restrict__ src) {
  const int4* s = reinterpret_cast<const int4*>(src);
  int4* d = reinterpret_cast<int4*>(dst);
  d[0] = s[0]; d[1] = s[1];
}
static __device__ __forceinline__ void zero32B(_Float16* dst) {
  int4 z; z.x = z.y = z.z = z.w = 0;
  int4* d = reinterpret_cast<int4*>(dst);
  d[0] = z; d[1] = z;
}

// CDNA5 async global->LDS copy (ASYNCcnt-tracked). Copies 32B per lane
// (two B128 ops; inst offset applies to both LDS and global address).
static __device__ __forceinline__ void async_cp32B(_Float16* lds, const _Float16* g) {
  unsigned l = (unsigned)(size_t)lds;   // low 32 bits of flat shared addr = LDS offset
  asm volatile("global_load_async_to_lds_b128 %0, %1, off\n\t"
               "global_load_async_to_lds_b128 %0, %1, off offset:16"
               :: "v"(l), "v"(g) : "memory");
}
static __device__ __forceinline__ void async_wait() {
  asm volatile("s_wait_asynccnt 0" ::: "memory");
}

// A fragment (16x32 f16) from a row-major tile, leading dim `ld` (elements).
// lanes 0-15 hold M=lane, K=0..15; lanes 16-31 M=lane-16, K=16..31 (matches load_B_T).
static __device__ __forceinline__ v16h load_A(const _Float16* base, int ld, int lane) {
  int m  = lane & 15;
  int kb = (lane >> 4) << 4;       // 0 or 16
  const _Float16* row = base + m * ld;
  v16h a;
#pragma unroll
  for (int e = 0; e < 16; ++e) a[e] = row[kb + e];        // K = kb + e
  return a;
}

// B fragment (32x16) where B[K][N] = rows[N][K]: per-lane contiguous K reads (2x b128).
// lanes 0-15: N=lane, K=0..15 ; lanes 16-31: N=lane-16, K=16..31.
static __device__ __forceinline__ v16h load_B_T(const _Float16* rows, int ld, int lane) {
  int n  = lane & 15;
  int kb = (lane >> 4) << 4;
  const _Float16* r = rows + n * ld + kb;
  v16h b;
#pragma unroll
  for (int e = 0; e < 16; ++e) b[e] = r[e];
  return b;
}

static __device__ __forceinline__ v8f wmma16(v16h a, v16h b, v8f c) {
  return __builtin_amdgcn_wmma_f32_16x16x32_f16(false, a, false, b, (short)0, c, false, false);
}

// row (0..MROWS-1) -> rolled-back output pixel index
static __device__ __forceinline__ long row_to_pixel(long row) {
  int w = (int)(row / NPIX), p = (int)(row % NPIX);
  int b = w >> 10, rem = w & 1023;
  int wy = rem >> 5, wx = rem & 31;
  int iy = p / 7, ix = p % 7;
  int oy = wy * 7 + iy + SHIFTX; if (oy >= IMG) oy -= IMG;
  int ox = wx * 7 + ix + SHIFTX; if (ox >= IMG) ox -= IMG;
  return (((long)b * IMG + oy) * IMG + ox);
}

// ---------------- kernels ----------------

// f32 weights -> f16 TRANSPOSED (WT[n][k] = W[k][n]) so GEMM B-tiles are K-contiguous.
__global__ void k_cvt_wT(const float* __restrict__ wq, const float* __restrict__ wk,
                         const float* __restrict__ wv, const float* __restrict__ wo,
                         _Float16* __restrict__ dst) {
  int i = blockIdx.x * blockDim.x + threadIdx.x;
  if (i >= 4 * CDIM * CDIM) return;
  int which = i / (CDIM * CDIM);
  int j = i % (CDIM * CDIM);
  int n = j / CDIM, k = j % CDIM;
  const float* src = which == 0 ? wq : which == 1 ? wk : which == 2 ? wv : wo;
  dst[i] = (_Float16)src[k * CDIM + n];
}

// Mask + shift-roll (folded into gather) + window partition + f16 pack.
__global__ void k_pre(const float* __restrict__ x, const float* __restrict__ y,
                      _Float16* __restrict__ xw, _Float16* __restrict__ yw,
                      float* __restrict__ lmask) {
  int wv = threadIdx.x >> 5, lane = threadIdx.x & 31;
  long pix = (long)blockIdx.x * 4 + wv;
  if (pix >= MROWS) return;
  long src = row_to_pixel(pix);          // same (i+shift)%IMG gather as the forward roll
  const float* xs = x + src * CDIM;
  const float* ys = y + src * CDIM;
  int c0 = lane * 6;
  float xv[6]; int cnt = 0;
#pragma unroll
  for (int i = 0; i < 6; ++i) { xv[i] = xs[c0 + i]; cnt += (xv[i] > 0.95f) ? 1 : 0; }
#pragma unroll
  for (int o = 16; o > 0; o >>= 1) cnt += __shfl_xor(cnt, o, 32);
  float m = (float)(CDIM - cnt) * (1.0f / (float)CDIM);
  _Float16* xd = xw + pix * CDIM + c0;
  _Float16* yd = yw + pix * CDIM + c0;
#pragma unroll
  for (int i = 0; i < 6; ++i) {
    xd[i] = (_Float16)(xv[i] * m);
    yd[i] = (_Float16)ys[c0 + i];
  }
  if (lane == 0) lmask[pix] = __logf(m + 1e-6f);
}

// C[M][192] = A[M][192] @ W[192][192] (WT passed transposed), f16 out.
// 128x64 tile per block, 4 waves, 8 WMMAs per wave per K-step.
__global__ void k_proj(const _Float16* __restrict__ A, const _Float16* __restrict__ WT,
                       _Float16* __restrict__ Out) {
  __shared__ alignas(16) _Float16 As[128][40];   // padded: 80B rows -> conflict-free b128
  __shared__ alignas(16) _Float16 BsT[64][40];
  __shared__ alignas(16) _Float16 Ds[128][72];
  int bx = blockIdx.x;
  int n0 = (bx % 3) * 64;
  long m0 = (long)(bx / 3) * 128;
  int t = threadIdx.x, lane = t & 31, wv = t >> 5;
  v8f acc[2][4];
#pragma unroll
  for (int i = 0; i < 2; ++i)
#pragma unroll
    for (int k = 0; k < 4; ++k)
#pragma unroll
      for (int j = 0; j < 8; ++j) acc[i][k][j] = 0.0f;

  for (int k0 = 0; k0 < CDIM; k0 += 32) {
    const _Float16* ar = A + (m0 + t) * CDIM + k0;
    async_cp32B(&As[t][0],  ar);
    async_cp32B(&As[t][16], ar + 16);
    if (t < 64) {
      const _Float16* br = WT + (long)(n0 + t) * CDIM + k0;
      async_cp32B(&BsT[t][0],  br);
      async_cp32B(&BsT[t][16], br + 16);
    }
    async_wait();
    __syncthreads();
    v16h a0 = load_A(&As[wv * 32][0],      40, lane);
    v16h a1 = load_A(&As[wv * 32 + 16][0], 40, lane);
#pragma unroll
    for (int nt = 0; nt < 4; ++nt) {
      v16h b = load_B_T(&BsT[nt * 16][0], 40, lane);
      acc[0][nt] = wmma16(a0, b, acc[0][nt]);
      acc[1][nt] = wmma16(a1, b, acc[1][nt]);
    }
    __syncthreads();
  }
  int n = lane & 15, rb = (lane >> 4) << 3;
#pragma unroll
  for (int mt = 0; mt < 2; ++mt)
#pragma unroll
    for (int nt = 0; nt < 4; ++nt)
#pragma unroll
      for (int r = 0; r < 8; ++r)
        Ds[wv * 32 + mt * 16 + rb + r][nt * 16 + n] = (_Float16)acc[mt][nt][r];
  __syncthreads();
  _Float16* dst = Out + (m0 + t) * CDIM + n0;     // coalesced 128b row writeback
#pragma unroll
  for (int i = 0; i < 4; ++i) cp32B(dst + i * 16, &Ds[t][i * 16]);
}

// Output projection: same GEMM, f32 result; window_reverse + roll folded into row mapping,
// D bounced through LDS so global stores are contiguous 256B rows.
__global__ void k_proj_o(const _Float16* __restrict__ A, const _Float16* __restrict__ WT,
                         float* __restrict__ Out) {
  __shared__ alignas(16) _Float16 As[128][40];
  __shared__ alignas(16) _Float16 BsT[64][40];
  __shared__ alignas(16) float    Dsf[128][68];
  int bx = blockIdx.x;
  int n0 = (bx % 3) * 64;
  long m0 = (long)(bx / 3) * 128;
  int t = threadIdx.x, lane = t & 31, wv = t >> 5;
  v8f acc[2][4];
#pragma unroll
  for (int i = 0; i < 2; ++i)
#pragma unroll
    for (int k = 0; k < 4; ++k)
#pragma unroll
      for (int j = 0; j < 8; ++j) acc[i][k][j] = 0.0f;

  for (int k0 = 0; k0 < CDIM; k0 += 32) {
    const _Float16* ar = A + (m0 + t) * CDIM + k0;
    async_cp32B(&As[t][0],  ar);
    async_cp32B(&As[t][16], ar + 16);
    if (t < 64) {
      const _Float16* br = WT + (long)(n0 + t) * CDIM + k0;
      async_cp32B(&BsT[t][0],  br);
      async_cp32B(&BsT[t][16], br + 16);
    }
    async_wait();
    __syncthreads();
    v16h a0 = load_A(&As[wv * 32][0],      40, lane);
    v16h a1 = load_A(&As[wv * 32 + 16][0], 40, lane);
#pragma unroll
    for (int nt = 0; nt < 4; ++nt) {
      v16h b = load_B_T(&BsT[nt * 16][0], 40, lane);
      acc[0][nt] = wmma16(a0, b, acc[0][nt]);
      acc[1][nt] = wmma16(a1, b, acc[1][nt]);
    }
    __syncthreads();
  }
  int n = lane & 15, rb = (lane >> 4) << 3;
#pragma unroll
  for (int mt = 0; mt < 2; ++mt)
#pragma unroll
    for (int nt = 0; nt < 4; ++nt)
#pragma unroll
      for (int r = 0; r < 8; ++r)
        Dsf[wv * 32 + mt * 16 + rb + r][nt * 16 + n] = acc[mt][nt][r];
  __syncthreads();
  long pxl = row_to_pixel(m0 + t);
  float* dst = Out + pxl * CDIM + n0;
  const int4* srow = reinterpret_cast<const int4*>(&Dsf[t][0]);
  int4* drow = reinterpret_cast<int4*>(dst);
#pragma unroll
  for (int i = 0; i < 16; ++i) drow[i] = srow[i];
}

// Fused attention per (window, head): QK^T -> +log-mask softmax -> AV.
// 49 padded to 64; d=32 == one WMMA K-step. V staged transposed so AV B-frags are contiguous.
__global__ void k_attn(const _Float16* __restrict__ q, const _Float16* __restrict__ kk,
                       const _Float16* __restrict__ vv, const float* __restrict__ lmaskg,
                       float* __restrict__ att_g, _Float16* __restrict__ outA) {
  __shared__ alignas(16) _Float16 qs[64][40];
  __shared__ alignas(16) _Float16 ks[64][40];
  __shared__ alignas(16) _Float16 vsT[32][72];   // [d][m]
  __shared__ float    lg[64][68];
  __shared__ alignas(16) _Float16 a16[64][72];
  __shared__ float    lm[64];

  int blk = blockIdx.x;
  int w = blk / NHEADS, hd = blk % NHEADS;
  int t = threadIdx.x, lane = t & 31, wv = t >> 5;

  const _Float16* qg = q  + ((long)w * NPIX) * CDIM + hd * DHEAD;
  const _Float16* kg = kk + ((long)w * NPIX) * CDIM + hd * DHEAD;
  const _Float16* vg = vv + ((long)w * NPIX) * CDIM + hd * DHEAD;

  { int row = t >> 1, seg = (t & 1) << 4;        // Q/K 64x32 tiles, zero-pad rows >= 49
    if (row < NPIX) {
      async_cp32B(&qs[row][seg], qg + (long)row * CDIM + seg);
      async_cp32B(&ks[row][seg], kg + (long)row * CDIM + seg);
    } else {
      zero32B(&qs[row][seg]); zero32B(&ks[row][seg]);
    }
  }
  { int m = t & 63, dseg = (t >> 6) << 4;        // V transposed into LDS
    _Float16 tmp[16];
    if (m < NPIX) {
      const int4* s = reinterpret_cast<const int4*>(vg + (long)m * CDIM + dseg);
      reinterpret_cast<int4*>(tmp)[0] = s[0];
      reinterpret_cast<int4*>(tmp)[1] = s[1];
    } else {
      int4 z; z.x = z.y = z.z = z.w = 0;
      reinterpret_cast<int4*>(tmp)[0] = z;
      reinterpret_cast<int4*>(tmp)[1] = z;
    }
#pragma unroll
    for (int i = 0; i < 16; ++i) vsT[dseg + i][m] = tmp[i];
  }
  if (t < 64) lm[t] = (t < NPIX) ? lmaskg[(long)w * NPIX + t] : 0.0f;
  async_wait();
  __syncthreads();

  // logits = scale * Q K^T + log(mask)   (B frag = contiguous-K read of row-major K tile)
  {
    v16h a = load_A(&qs[wv * 16][0], 40, lane);
    int n = lane & 15, rb = (lane >> 4) << 3;
#pragma unroll
    for (int nt = 0; nt < 4; ++nt) {
      v16h b = load_B_T(&ks[nt * 16][0], 40, lane);
      v8f c;
#pragma unroll
      for (int j = 0; j < 8; ++j) c[j] = 0.0f;
      c = wmma16(a, b, c);
      int col = nt * 16 + n;
      float add = lm[col];
#pragma unroll
      for (int r = 0; r < 8; ++r)
        lg[wv * 16 + rb + r][col] = c[r] * 0.1767766952966369f + add;  // 1/sqrt(32)
    }
  }
  __syncthreads();

  // row softmax over keys 0..48; emit f32 att + f16 probs (zero-padded to 64)
  if (t < 64) {
    int row = t;
    if (row < NPIX) {
      float mx = -1e30f;
      for (int m = 0; m < NPIX; ++m) mx = fmaxf(mx, lg[row][m]);
      float s = 0.0f;
      for (int m = 0; m < NPIX; ++m) { float e = __expf(lg[row][m] - mx); lg[row][m] = e; s += e; }
      float inv = 1.0f / s;
      float* ag = att_g + (((long)w * NHEADS + hd) * NPIX + row) * NPIX;
      for (int m = 0; m < NPIX; ++m) {
        float pv = lg[row][m] * inv;
        a16[row][m] = (_Float16)pv;
        ag[m] = pv;
      }
      for (int m = NPIX; m < 64; ++m) a16[row][m] = (_Float16)0.0f;
    } else {
      for (int m = 0; m < 64; ++m) a16[row][m] = (_Float16)0.0f;
    }
  }
  __syncthreads();

  // out = att @ V   (K = 64 padded keys -> 2 WMMA steps; d = 32 -> 2 N-tiles)
  {
    v16h a0 = load_A(&a16[wv * 16][0],  72, lane);
    v16h a1 = load_A(&a16[wv * 16][32], 72, lane);
    int n = lane & 15, rb = (lane >> 4) << 3;
#pragma unroll
    for (int dt = 0; dt < 2; ++dt) {
      v8f c;
#pragma unroll
      for (int j = 0; j < 8; ++j) c[j] = 0.0f;
      v16h b0 = load_B_T(&vsT[dt * 16][0],  72, lane);   // K = 0..31
      c = wmma16(a0, b0, c);
      v16h b1 = load_B_T(&vsT[dt * 16][32], 72, lane);   // K = 32..63
      c = wmma16(a1, b1, c);
#pragma unroll
      for (int r = 0; r < 8; ++r)
        qs[wv * 16 + rb + r][dt * 16 + n] = (_Float16)c[r];   // qs reused as D staging
    }
  }
  __syncthreads();
  { int row = t >> 1, seg = (t & 1) << 4;       // coalesced writeback of rows < 49
    if (row < NPIX)
      cp32B(outA + ((long)w * NPIX + row) * CDIM + hd * DHEAD + seg, &qs[row][seg]);
  }
}

// ---------------- launch ----------------

extern "C" void kernel_launch(void* const* d_in, const int* in_sizes, int n_in,
                              void* d_out, int out_size, void* d_ws, size_t ws_size,
                              hipStream_t stream) {
  (void)in_sizes; (void)n_in; (void)out_size; (void)ws_size;
  const float* x  = (const float*)d_in[0];
  const float* y  = (const float*)d_in[1];
  const float* Wq = (const float*)d_in[2];
  const float* Wk = (const float*)d_in[3];
  const float* Wv = (const float*)d_in[4];
  const float* Wo = (const float*)d_in[5];
  float* out = (float*)d_out;
  float* att = out + (size_t)OUT_ELEMS;

  // workspace layout (~590 MB):
  //   [0, 294912)          f16 transposed weights WqT|WkT|WvT|WoT
  //   [294912, +1605632)   f32 log-mask per token row
  //   4 big f16 buffers MROWS*CDIM with reuse:
  //     bufA: xw -> V    bufB: yw -> attn-out    bufC: Q    bufD: K
  char* ws = (char*)d_ws;
  _Float16* w16  = (_Float16*)ws;
  float* lmask   = (float*)(ws + 294912);
  size_t big0    = 294912 + 1605632;
  size_t BIG     = (size_t)MROWS * CDIM * sizeof(_Float16);
  _Float16* bufA = (_Float16*)(ws + big0);
  _Float16* bufB = (_Float16*)(ws + big0 + BIG);
  _Float16* bufC = (_Float16*)(ws + big0 + 2 * BIG);
  _Float16* bufD = (_Float16*)(ws + big0 + 3 * BIG);
  _Float16* wqT = w16;
  _Float16* wkT = w16 + 36864;
  _Float16* wvT = w16 + 2 * 36864;
  _Float16* woT = w16 + 3 * 36864;

  k_cvt_wT<<<(4 * CDIM * CDIM + 255) / 256, 256, 0, stream>>>(Wq, Wk, Wv, Wo, w16);
  k_pre<<<MROWS / 4, 128, 0, stream>>>(x, y, bufA, bufB, lmask);

  dim3 gp((MROWS / 128) * 3);                      // 3136 * 3 tiles
  k_proj<<<gp, 128, 0, stream>>>(bufA, wqT, bufC);         // Q = xw @ Wq
  k_proj<<<gp, 128, 0, stream>>>(bufB, wkT, bufD);         // K = yw @ Wk
  k_proj<<<gp, 128, 0, stream>>>(bufB, wvT, bufA);         // V = yw @ Wv (xw dead)

  k_attn<<<NWIN * NHEADS, 128, 0, stream>>>(bufC, bufD, bufA, lmask, att, bufB);

  k_proj_o<<<gp, 128, 0, stream>>>(bufB, woT, out);        // out = attn @ Wo (+reverse/roll)
}
